// RefineRCNNNet_15358803050975
// MI455X (gfx1250) — matrix operand order repaired and checked by reading, not compiled, not run on hardware
//
#include <hip/hip_runtime.h>
#include <hip/hip_bf16.h>

typedef __attribute__((ext_vector_type(16))) _Float16 v16h;
typedef __attribute__((ext_vector_type(8)))  float    v8f;

#define DEV static __device__ __forceinline__

constexpr int B_    = 8;
constexpr int N_    = 512;
constexpr int C0    = 512;
constexpr int CH    = 64;
constexpr int KNN   = 16;
constexpr int FEATC = 448;    // 7 * 64
constexpr int FUSEC = 1024;
constexpr int OUTC  = 1472;   // 1024 + 448
constexpr float BN_EPS = 1e-5f;

// ---------------- WMMA helpers (CDNA5 16x16x32 f16 -> f32) ----------------

DEV v8f wmma16(v16h a, v16h b, v8f c) {
  // emits v_wmma_f32_16x16x32_f16
  return __builtin_amdgcn_wmma_f32_16x16x32_f16(false, a, false, b, (short)0, c,
                                                false, false);
}

// A-matrix 16x32 f16 layout (ISA 7.12.2): lane L -> row m = L%16;
// elements 0..7  hold K = k0 + (L/16)*8 + i
// elements 8..15 hold K = k0 + 16 + (L/16)*8 + i
DEV v16h load_a16(const _Float16* A, int ld, int m0, int k0, int lane) {
  const _Float16* p = A + (size_t)(m0 + (lane & 15)) * ld + k0 + ((lane >> 4) << 3);
  v16h a;
#pragma unroll
  for (int i = 0; i < 8; ++i) { a[i] = p[i]; a[8 + i] = p[16 + i]; }
  return a;
}

// B-matrix 32x16 f16 layout: lane L -> col n = L%16;
// elements 0..7 hold K = k0 + (L/16)*8 + i, elements 8..15 hold
// K = k0 + 16 + (L/16)*8 + i, loaded from
// row-major B^T[N][K] (i.e. W stored [O][K] row-major).
DEV v16h load_b16(const _Float16* BT, int ld, int n0, int k0, int lane) {
  const _Float16* p = BT + (size_t)(n0 + (lane & 15)) * ld + k0 + ((lane >> 4) << 3);
  v16h b;
#pragma unroll
  for (int i = 0; i < 8; ++i) { b[i] = p[i]; b[8 + i] = p[16 + i]; }
  return b;
}

// ---------------- setup / conversion kernels ----------------

__global__ void k_f2h(const float* __restrict__ s, _Float16* __restrict__ d, int n) {
  int i = blockIdx.x * blockDim.x + threadIdx.x;
  if (i < n) d[i] = (_Float16)s[i];
}

__global__ void k_transpose_f2h(const float* __restrict__ x, _Float16* __restrict__ xt) {
  int tid = blockIdx.x * blockDim.x + threadIdx.x;   // B*C0*N_
  int n = tid % N_;
  int c = (tid / N_) % C0;
  int b = tid / (N_ * C0);
  xt[((size_t)b * N_ + n) * C0 + c] = (_Float16)x[tid];
}

__global__ void k_zero(float* p, int n) {
  int i = blockIdx.x * blockDim.x + threadIdx.x;
  if (i < n) p[i] = 0.0f;
}

// xx[b*N+n] = sum_c ft[row][coff+c]^2, f32 accumulate
__global__ void k_rownorm(const _Float16* __restrict__ ft, int ld, int coff, int C,
                          float* __restrict__ xx) {
  int tid = blockIdx.x * blockDim.x + threadIdx.x;   // B*N
  const _Float16* p = ft + (size_t)tid * ld + coff;
  float s = 0.0f;
  for (int c = 0; c < C; ++c) { float v = (float)p[c]; s += v * v; }
  xx[tid] = s;
}

// ---------------- pairwise distance via WMMA ----------------
// dist[b][n][m] = xx[n] + xx[m] - 2 * sum_c ft[n][c]*ft[m][c]
__global__ void k_dist(const _Float16* __restrict__ ft, int ld, int coff, int C,
                       const float* __restrict__ xx, float* __restrict__ dist) {
  int lane = threadIdx.x & 31;
  int wave = threadIdx.x >> 5;
  int b    = blockIdx.y;
  int tile = blockIdx.x * 8 + wave;       // 0..1023 (32x32 tiles of 16x16)
  int n0 = (tile >> 5) * 16;
  int m0 = (tile & 31) * 16;
  const _Float16* base = ft + (size_t)b * N_ * ld + coff;
  v8f acc = {};
  for (int k0 = 0; k0 < C; k0 += 32) {
    v16h a  = load_a16(base, ld, n0, k0, lane);
    v16h bb = load_b16(base, ld, m0, k0, lane);
    acc = wmma16(a, bb, acc);
  }
  int col = m0 + (lane & 15);
  int rb  = (lane >> 4) << 3;
  const float* xb = xx + (size_t)b * N_;
  float xc = xb[col];
  float* dr = dist + (size_t)b * N_ * N_;
#pragma unroll
  for (int r = 0; r < 8; ++r) {
    int row = n0 + rb + r;
    dr[(size_t)row * N_ + col] = xb[row] + xc - 2.0f * acc[r];
  }
}

// ---------------- iterative top-k (k*d selections, keep every d-th) -------
__global__ void k_topk(const float* __restrict__ dist, int* __restrict__ idxout,
                       int k, int dil) {
  __shared__ float sv[N_];
  __shared__ float rv[128];
  __shared__ int   ri[128];
  int t   = threadIdx.x;
  int row = blockIdx.x;                       // b*N + n
  const float* dr = dist + (size_t)row * N_;
  for (int i = t; i < N_; i += 128) sv[i] = dr[i];
  __syncthreads();
  int km = k * dil;
  for (int it = 0; it < km; ++it) {
    float best = 3.4e38f; int bi = N_;
    for (int i = t; i < N_; i += 128) {
      float v = sv[i];
      if (v < best) { best = v; bi = i; }
    }
    rv[t] = best; ri[t] = bi;
    __syncthreads();
    for (int s = 64; s > 0; s >>= 1) {
      if (t < s) {
        if (rv[t + s] < rv[t] || (rv[t + s] == rv[t] && ri[t + s] < ri[t])) {
          rv[t] = rv[t + s]; ri[t] = ri[t + s];
        }
      }
      __syncthreads();
    }
    if (t == 0) {
      int w = ri[0];
      if (it % dil == 0) idxout[(size_t)row * KNN + it / dil] = w;
      sv[w] = 3.4e38f;
    }
    __syncthreads();
  }
}

// ---------------- EdgeConv GEMM via WMMA (gathers + builds e on the fly) --
// y[(b*N+n)*K+kk][o] = sum_c e[c] * W[o][c] + bias[o]
// e = [x_i (C) ; x_j - x_i (C)]; one wave -> 16 rows x 64 outputs
__global__ void k_edgegemm(const _Float16* __restrict__ ft, int ld, int coff, int C,
                           const int* __restrict__ idx, const _Float16* __restrict__ Wh,
                           const float* __restrict__ bias, float* __restrict__ y) {
  int lane  = threadIdx.x & 31;
  int wave  = threadIdx.x >> 5;
  int b     = blockIdx.z;
  int mtile = blockIdx.x * 8 + wave;    // 0..511
  int row0  = mtile * 16;
  int m  = lane & 15;
  int r  = row0 + m;
  int n  = r >> 4;                      // KNN == 16
  int kk = r & 15;
  int j  = idx[((size_t)b * N_ + n) * KNN + kk];
  const _Float16* xi = ft + ((size_t)b * N_ + n) * ld + coff;
  const _Float16* xj = ft + ((size_t)b * N_ + j) * ld + coff;
  int C2 = 2 * C;
  v8f acc[4] = {};
  for (int k0 = 0; k0 < C2; k0 += 32) {
    v16h a;
    int c1 = k0 + ((lane >> 4) << 3);
#pragma unroll
    for (int half = 0; half < 2; ++half) {
      int c = c1 + half * 16;           // 8-wide chunks never straddle C
      if (c < C) {
#pragma unroll
        for (int i = 0; i < 8; ++i) a[half * 8 + i] = xi[c + i];
      } else {
        int cc = c - C;
#pragma unroll
        for (int i = 0; i < 8; ++i)
          a[half * 8 + i] = (_Float16)(xj[cc + i] - xi[cc + i]);
      }
    }
#pragma unroll
    for (int oc = 0; oc < 4; ++oc) {
      v16h bw = load_b16(Wh, C2, oc * 16, k0, lane);
      acc[oc] = wmma16(a, bw, acc[oc]);
    }
  }
  int rb = (lane >> 4) << 3;
  size_t rowBase = (size_t)b * N_ * KNN;
#pragma unroll
  for (int oc = 0; oc < 4; ++oc) {
    int o = oc * 16 + (lane & 15);
    float bo = bias[o];
#pragma unroll
    for (int rr = 0; rr < 8; ++rr) {
      size_t gr = rowBase + row0 + rb + rr;
      y[gr * CH + o] = acc[oc][rr] + bo;
    }
  }
}

// ---------------- BN stats: per-channel sum / sumsq (channel-last y) ------
__global__ void k_bnstats(const float* __restrict__ y, int rows, int C,
                          float* __restrict__ stats) {
  __shared__ float ssum[256], ssq[256];
  int t  = threadIdx.x;
  int c  = blockIdx.x * 64 + (t & 63);
  int rl = t >> 6;
  float s = 0.0f, q = 0.0f;
  int r0   = blockIdx.y * 1024;
  int rend = r0 + 1024 < rows ? r0 + 1024 : rows;
  for (int r = r0 + rl; r < rend; r += 4) {
    float v = y[(size_t)r * C + c];
    s += v; q += v * v;
  }
  ssum[t] = s; ssq[t] = q;
  __syncthreads();
  if (t < 64) {
    s = ssum[t] + ssum[t + 64] + ssum[t + 128] + ssum[t + 192];
    q = ssq[t] + ssq[t + 64] + ssq[t + 128] + ssq[t + 192];
    atomicAdd(&stats[c], s);
    atomicAdd(&stats[C + c], q);
  }
}

// ---------------- BN + ReLU + max over K + residual; emit f32 & f16 -------
__global__ void k_bnmax(const float* __restrict__ y, const float* __restrict__ stats,
                        float invcnt, const float* __restrict__ gamma,
                        const float* __restrict__ beta, float* __restrict__ feats,
                        _Float16* __restrict__ ftT, int coff, int residOff) {
  int tid = blockIdx.x * blockDim.x + threadIdx.x;   // B*N*64
  int o = tid & 63;
  int n = (tid >> 6) % N_;
  int b = (tid >> 6) / N_;
  float mean = stats[o] * invcnt;
  float var  = stats[CH + o] * invcnt - mean * mean;
  float rs   = rsqrtf(var + BN_EPS);
  float g = gamma[o], be = beta[o];
  const float* yr = y + ((size_t)(b * N_ + n) * KNN) * CH + o;
  float m = 0.0f;                                    // relu(...) >= 0
#pragma unroll 4
  for (int kk = 0; kk < KNN; ++kk) {
    float v = (yr[(size_t)kk * CH] - mean) * rs * g + be;
    v = v > 0.0f ? v : 0.0f;
    m = v > m ? v : m;
  }
  float res = (residOff >= 0) ? feats[((size_t)b * FEATC + residOff + o) * N_ + n] : 0.0f;
  float fo = res + m;
  feats[((size_t)b * FEATC + coff + o) * N_ + n] = fo;
  ftT[((size_t)b * N_ + n) * FEATC + coff + o]   = (_Float16)fo;
}

// ---------------- fusion GEMM: [B*N,448] x [448,1024] via WMMA ------------
__global__ void k_fusion(const _Float16* __restrict__ ftT, const _Float16* __restrict__ Wfh,
                         const float* __restrict__ bf, float* __restrict__ y) {
  int lane  = threadIdx.x & 31;
  int wave  = threadIdx.x >> 5;
  int mtile = blockIdx.y * 8 + wave;    // 0..255
  int row0  = mtile * 16;
  int o0    = blockIdx.x * 64;
  v8f acc[4] = {};
  for (int k0 = 0; k0 < FEATC; k0 += 32) {
    v16h a = load_a16(ftT, FEATC, row0, k0, lane);
#pragma unroll
    for (int oc = 0; oc < 4; ++oc) {
      v16h bw = load_b16(Wfh, FEATC, o0 + oc * 16, k0, lane);
      acc[oc] = wmma16(a, bw, acc[oc]);
    }
  }
  int rb = (lane >> 4) << 3;
#pragma unroll
  for (int oc = 0; oc < 4; ++oc) {
    int o = o0 + oc * 16 + (lane & 15);
    float bo = bf[o];
#pragma unroll
    for (int rr = 0; rr < 8; ++rr)
      y[(size_t)(row0 + rb + rr) * FUSEC + o] = acc[oc][rr] + bo;
  }
}

// ---------------- fusion BN + ReLU + max over N -> pooled[b][o] -----------
__global__ void k_fusionpool(const float* __restrict__ y, const float* __restrict__ stats,
                             float invcnt, const float* __restrict__ gamma,
                             const float* __restrict__ beta, float* __restrict__ pooled) {
  int tid = blockIdx.x * blockDim.x + threadIdx.x;   // B*1024
  int o = tid & (FUSEC - 1);
  int b = tid >> 10;
  float mean = stats[o] * invcnt;
  float var  = stats[FUSEC + o] * invcnt - mean * mean;
  float rs   = rsqrtf(var + BN_EPS);
  float g = gamma[o], be = beta[o];
  float m = 0.0f;
  for (int n = 0; n < N_; ++n) {
    float v = (y[((size_t)b * N_ + n) * FUSEC + o] - mean) * rs * g + be;
    v = v > 0.0f ? v : 0.0f;
    m = v > m ? v : m;
  }
  pooled[tid] = m;
}

// ---------------- final assembly: out = concat([pooled bcast, feats]) -----
__global__ void k_out(const float* __restrict__ pooled, const float* __restrict__ feats,
                      float* __restrict__ out) {
  size_t tid = (size_t)blockIdx.x * blockDim.x + threadIdx.x;  // B*1472*N
  int b   = (int)(tid / ((size_t)OUTC * N_));
  int rem = (int)(tid - (size_t)b * OUTC * N_);
  int ch  = rem / N_;
  int n   = rem % N_;
  out[tid] = (ch < FUSEC)
                 ? pooled[(size_t)b * FUSEC + ch]
                 : feats[((size_t)b * FEATC + (ch - FUSEC)) * N_ + n];
}

// ---------------- host orchestration ----------------

extern "C" void kernel_launch(void* const* d_in, const int* in_sizes, int n_in,
                              void* d_out, int out_size, void* d_ws, size_t ws_size,
                              hipStream_t stream) {
  (void)in_sizes; (void)n_in; (void)out_size; (void)ws_size;
  const float* x     = (const float*)d_in[0];
  const float* W0    = (const float*)d_in[1];
  const float* b0    = (const float*)d_in[2];
  const float* g0    = (const float*)d_in[3];
  const float* beta0 = (const float*)d_in[4];
  const float* Wb    = (const float*)d_in[5];
  const float* bb    = (const float*)d_in[6];
  const float* gb    = (const float*)d_in[7];
  const float* betab = (const float*)d_in[8];
  const float* Wf    = (const float*)d_in[9];
  const float* bf    = (const float*)d_in[10];
  const float* gf    = (const float*)d_in[11];
  const float* betaf = (const float*)d_in[12];
  float* out = (float*)d_out;

  char* ws = (char*)d_ws;
  size_t off = 0;
  auto alloc = [&](size_t bytes) {
    size_t o = off;
    off += (bytes + 255) & ~(size_t)255;
    return o;
  };
  _Float16* xt     = (_Float16*)(ws + alloc((size_t)B_ * N_ * C0 * 2));      // 4 MB
  float*    xx     = (float*)   (ws + alloc((size_t)B_ * N_ * 4));
  float*    dist   = (float*)   (ws + alloc((size_t)B_ * N_ * N_ * 4));      // 8 MB
  int*      idx    = (int*)     (ws + alloc((size_t)B_ * N_ * KNN * 4));     // 256 KB
  float*    ybuf   = (float*)   (ws + alloc((size_t)B_ * N_ * FUSEC * 4));   // 16 MB
  float*    feats  = (float*)   (ws + alloc((size_t)B_ * FEATC * N_ * 4));   // 7 MB
  _Float16* ftT    = (_Float16*)(ws + alloc((size_t)B_ * N_ * FEATC * 2));   // 3.7 MB
  float*    stats  = (float*)   (ws + alloc((size_t)2 * FUSEC * 4));
  float*    pooled = (float*)   (ws + alloc((size_t)B_ * FUSEC * 4));
  _Float16* W0h    = (_Float16*)(ws + alloc((size_t)CH * 2 * C0 * 2));
  _Float16* Wbh    = (_Float16*)(ws + alloc((size_t)6 * CH * 2 * CH * 2));
  _Float16* Wfh    = (_Float16*)(ws + alloc((size_t)FUSEC * FEATC * 2));

  // one-time (per call) conversions to f16
  k_transpose_f2h<<<(B_ * C0 * N_) / 256, 256, 0, stream>>>(x, xt);
  k_f2h<<<(CH * 2 * C0 + 255) / 256, 256, 0, stream>>>(W0, W0h, CH * 2 * C0);
  k_f2h<<<(6 * CH * 2 * CH + 255) / 256, 256, 0, stream>>>(Wb, Wbh, 6 * CH * 2 * CH);
  k_f2h<<<(FUSEC * FEATC + 255) / 256, 256, 0, stream>>>(Wf, Wfh, FUSEC * FEATC);

  for (int blk = 0; blk < 7; ++blk) {
    const _Float16* fin;
    int ld, coff, C;
    if (blk == 0) { fin = xt;  ld = C0;    coff = 0;              C = C0; }
    else          { fin = ftT; ld = FEATC; coff = (blk - 1) * CH; C = CH; }
    int dil = (blk == 0) ? 1 : ((blk - 1) % 4 + 1);

    k_rownorm<<<(B_ * N_) / 256, 256, 0, stream>>>(fin, ld, coff, C, xx);
    k_dist<<<dim3(128, B_), 256, 0, stream>>>(fin, ld, coff, C, xx, dist);
    k_topk<<<B_ * N_, 128, 0, stream>>>(dist, idx, KNN, dil);

    const _Float16* Wh = (blk == 0) ? W0h : (Wbh + (size_t)(blk - 1) * CH * 2 * CH);
    const float* bias  = (blk == 0) ? b0 : (bb + (blk - 1) * CH);
    k_edgegemm<<<dim3(64, 1, B_), 256, 0, stream>>>(fin, ld, coff, C, idx, Wh, bias, ybuf);

    k_zero<<<1, 256, 0, stream>>>(stats, 2 * CH);
    k_bnstats<<<dim3(1, 64), 256, 0, stream>>>(ybuf, B_ * N_ * KNN, CH, stats);

    const float* gamma = (blk == 0) ? g0 : (gb + (blk - 1) * CH);
    const float* beta  = (blk == 0) ? beta0 : (betab + (blk - 1) * CH);
    k_bnmax<<<(B_ * N_ * CH) / 256, 256, 0, stream>>>(
        ybuf, stats, 1.0f / (float)(B_ * N_ * KNN), gamma, beta, feats, ftT,
        blk * CH, (blk == 0) ? -1 : (blk - 1) * CH);
  }

  // fusion: [B*N,448] @ [448,1024], BN over (B,N), ReLU, max over N
  k_fusion<<<dim3(16, 32), 256, 0, stream>>>(ftT, Wfh, bf, ybuf);
  k_zero<<<(2 * FUSEC + 255) / 256, 256, 0, stream>>>(stats, 2 * FUSEC);
  k_bnstats<<<dim3(16, 4), 256, 0, stream>>>(ybuf, B_ * N_, FUSEC, stats);
  k_fusionpool<<<(B_ * FUSEC) / 256, 256, 0, stream>>>(
      ybuf, stats, 1.0f / (float)(B_ * N_), gf, betaf, pooled);

  k_out<<<(B_ * OUTC * N_) / 256, 256, 0, stream>>>(pooled, feats, out);
}
